// EdgeDecoder_11768210391490
// MI455X (gfx1250) — matrix operand order repaired and checked by reading, not compiled, not run on hardware
//
#include <hip/hip_runtime.h>

typedef __attribute__((ext_vector_type(16))) __bf16 v16bf;
typedef __attribute__((ext_vector_type(8)))  float  v8f;

#define WAVES_PER_BLOCK 8
#define THREADS (WAVES_PER_BLOCK * 32)

__device__ __forceinline__ void cvt_frag(const float4 gi[4], v16bf& a) {
    a[0]  = (__bf16)gi[0].x; a[1]  = (__bf16)gi[0].y;
    a[2]  = (__bf16)gi[0].z; a[3]  = (__bf16)gi[0].w;
    a[4]  = (__bf16)gi[1].x; a[5]  = (__bf16)gi[1].y;
    a[6]  = (__bf16)gi[1].z; a[7]  = (__bf16)gi[1].w;
    a[8]  = (__bf16)gi[2].x; a[9]  = (__bf16)gi[2].y;
    a[10] = (__bf16)gi[2].z; a[11] = (__bf16)gi[2].w;
    a[12] = (__bf16)gi[3].x; a[13] = (__bf16)gi[3].y;
    a[14] = (__bf16)gi[3].z; a[15] = (__bf16)gi[3].w;
}

__device__ __forceinline__ void load_chunks(const float* nb, int chunkOff,
                                            float4 go[4]) {
    const float4* q0 = (const float4*)(nb + chunkOff);
    const float4* q1 = (const float4*)(nb + 16 + chunkOff);
    go[0] = q0[0]; go[1] = q0[1]; go[2] = q1[0]; go[3] = q1[1];
}

// One k-step for a 32-edge tile: convert 2 A fragments, prefetch next step's
// A chunks into the other buffer, then 8 B fragments x 2 WMMAs each.
__device__ __forceinline__ void mma_stage32(
    int ks, int kn, bool doPf,
    const v16bf* __restrict__ laneFrag,
    const float* __restrict__ srcLo, const float* __restrict__ dstLo,
    const float* __restrict__ srcHi, const float* __restrict__ dstHi,
    int chunkOff,
    const float4 giLo[4], const float4 giHi[4],
    float4 goLo[4], float4 goHi[4],
    v8f cLo[8], v8f cHi[8])
{
    v16bf aLo, aHi;
    cvt_frag(giLo, aLo);
    cvt_frag(giHi, aHi);

    if (doPf) {
        const float* nbLo = (kn < 4) ? (srcLo + kn * 32) : (dstLo + (kn - 4) * 32);
        const float* nbHi = (kn < 4) ? (srcHi + kn * 32) : (dstHi + (kn - 4) * 32);
        load_chunks(nbLo, chunkOff, goLo);
        load_chunks(nbHi, chunkOff, goHi);
    }

    const v16bf* fragBase = laneFrag + ks * 8 * 32;
    v16bf bcur = fragBase[0];
    #pragma unroll
    for (int nt = 0; nt < 8; ++nt) {
        v16bf bnxt = bcur;
        if (nt < 7) bnxt = fragBase[(nt + 1) * 32];
        cLo[nt] = __builtin_amdgcn_wmma_f32_16x16x32_bf16(
            false, aLo, false, bcur, (short)0, cLo[nt], false, false);
        cHi[nt] = __builtin_amdgcn_wmma_f32_16x16x32_bf16(
            false, aHi, false, bcur, (short)0, cHi[nt], false, false);
        bcur = bnxt;
    }
}

__global__ __launch_bounds__(THREADS)
void edge_decoder_wmma_kernel(const float* __restrict__ z_src,
                              const float* __restrict__ z_dst,
                              const int*   __restrict__ eli,
                              const float* __restrict__ W1,
                              const float* __restrict__ b1,
                              const float* __restrict__ W2,
                              const float* __restrict__ b2,
                              float*       __restrict__ out,
                              int E, int nTiles)
{
    // 64 B-fragments (8 k-steps x 8 n-tiles) x 32 lanes x 32B = 64 KB
    __shared__ v16bf ldsB[64 * 32];

    const int tid = threadIdx.x;

    // ---- stage W1 (256x128 f32, [k][n]) into LDS as bf16, pre-swizzled ----
    unsigned int* ldsBu = (unsigned int*)ldsB;
    #pragma unroll 8
    for (int i = 0; i < 16384 / THREADS; ++i) {
        int idx = tid + i * THREADS;
        int f   = idx >> 8;
        int rem = idx & 255;
        int l   = rem >> 3;
        int d   = rem & 7;
        int ks  = f >> 3;
        int nt  = f & 7;
        int hfl = l >> 4;
        int nL  = l & 15;
        int k   = ks * 32 + (d >> 2) * 16 + hfl * 8 + 2 * (d & 3);
        int n   = nt * 16 + nL;
        __bf16 blo = (__bf16)W1[k * 128 + n];
        __bf16 bhi = (__bf16)W1[(k + 1) * 128 + n];
        unsigned int packed =
            ((unsigned int)__builtin_bit_cast(unsigned short, bhi) << 16) |
             (unsigned int)__builtin_bit_cast(unsigned short, blo);
        ldsBu[idx] = packed;
    }
    __syncthreads();

    const int lane   = tid & 31;
    const int waveId = tid >> 5;
    const int hf     = lane >> 4;
    const int m      = lane & 15;

    float b1r[8], w2r[8];
    #pragma unroll
    for (int nt = 0; nt < 8; ++nt) {
        int n = m + nt * 16;
        b1r[nt] = b1[n];
        w2r[nt] = W2[n];
    }
    const float b2v = b2[0];

    const int gw  = blockIdx.x * WAVES_PER_BLOCK + waveId;
    const int gws = gridDim.x * WAVES_PER_BLOCK;
    const int chunkOff = hf * 8;
    const v16bf* laneFrag = &ldsB[lane];

    // ---- pipeline prologue: indices + ks=0 A chunks for the first tile ----
    int rLoN, cLoN, rHiN, cHiN;
    float4 gLo[4], gHi[4], hLo[4], hHi[4];
    {
        int t0  = min(gw, nTiles - 1);
        int eLo = min(t0 * 32 + m,      E - 1);
        int eHi = min(t0 * 32 + 16 + m, E - 1);
        rLoN = eli[eLo]; cLoN = eli[E + eLo];
        rHiN = eli[eHi]; cHiN = eli[E + eHi];
        load_chunks(z_src + (long)rLoN * 128, chunkOff, gLo);
        load_chunks(z_src + (long)rHiN * 128, chunkOff, gHi);
    }
    #pragma unroll
    for (int i = 0; i < 4; ++i) { hLo[i] = gLo[i]; hHi[i] = gHi[i]; }

    for (int t = gw; t < nTiles; t += gws) {
        const int e0 = t * 32;
        const float* srcLo = z_src + (long)rLoN * 128;
        const float* dstLo = z_dst + (long)cLoN * 128;
        const float* srcHi = z_src + (long)rHiN * 128;
        const float* dstHi = z_dst + (long)cHiN * 128;

        // prefetch NEXT tile's indices (hidden under the WMMA phase)
        {
            int tn  = min(t + gws, nTiles - 1);
            int eLo = min(tn * 32 + m,      E - 1);
            int eHi = min(tn * 32 + 16 + m, E - 1);
            rLoN = eli[eLo]; cLoN = eli[E + eLo];
            rHiN = eli[eHi]; cHiN = eli[E + eHi];
        }

        v8f cLo[8], cHi[8];
        #pragma unroll
        for (int nt = 0; nt < 8; ++nt) { cLo[nt] = (v8f)0.0f; cHi[nt] = (v8f)0.0f; }

        // real loop (no hoist/spill of B), manually 2x unrolled double-buffer
        #pragma unroll 1
        for (int ks = 0; ks < 8; ks += 2) {
            mma_stage32(ks,     ks + 1, true,
                        laneFrag, srcLo, dstLo, srcHi, dstHi, chunkOff,
                        gLo, gHi, hLo, hHi, cLo, cHi);
            mma_stage32(ks + 1, ks + 2, (ks + 2) < 8,
                        laneFrag, srcLo, dstLo, srcHi, dstHi, chunkOff,
                        hLo, hHi, gLo, gHi, cLo, cHi);
        }

        // issue NEXT tile's ks=0 A loads; epilogue VALU hides the latency
        load_chunks(z_src + (long)rLoN * 128, chunkOff, gLo);
        load_chunks(z_src + (long)rHiN * 128, chunkOff, gHi);

        // epilogue: relu(D + b1) dot W2, butterfly-reduce across N, store 32.
        float accLo[8], accHi[8];
        #pragma unroll
        for (int r = 0; r < 8; ++r) { accLo[r] = 0.0f; accHi[r] = 0.0f; }
        #pragma unroll
        for (int nt = 0; nt < 8; ++nt) {
            #pragma unroll
            for (int r = 0; r < 8; ++r) {
                float hl = fmaxf(cLo[nt][r] + b1r[nt], 0.0f);
                float hh = fmaxf(cHi[nt][r] + b1r[nt], 0.0f);
                accLo[r] = fmaf(hl, w2r[nt], accLo[r]);
                accHi[r] = fmaf(hh, w2r[nt], accHi[r]);
            }
        }
        #pragma unroll
        for (int off = 8; off >= 1; off >>= 1) {
            #pragma unroll
            for (int r = 0; r < 8; ++r) {
                accLo[r] += __shfl_xor(accLo[r], off, 32);
                accHi[r] += __shfl_xor(accHi[r], off, 32);
            }
        }
        if (m == 0) {
            #pragma unroll
            for (int r = 0; r < 8; ++r) {
                int eo = e0 + hf * 8 + r;
                if (eo < E)      out[eo]      = accLo[r] + b2v;
                if (eo + 16 < E) out[eo + 16] = accHi[r] + b2v;
            }
        }
    }
}

extern "C" void kernel_launch(void* const* d_in, const int* in_sizes, int n_in,
                              void* d_out, int out_size, void* d_ws, size_t ws_size,
                              hipStream_t stream) {
    const float* z_src = (const float*)d_in[0];
    const float* z_dst = (const float*)d_in[1];
    const int*   eli   = (const int*)d_in[2];
    const float* W1    = (const float*)d_in[3];
    const float* b1    = (const float*)d_in[4];
    const float* W2    = (const float*)d_in[5];
    const float* b2    = (const float*)d_in[6];
    float* out = (float*)d_out;

    const int E = in_sizes[2] / 2;
    const int nTiles = (E + 31) / 32;
    int blocks = (nTiles + WAVES_PER_BLOCK - 1) / WAVES_PER_BLOCK;
    if (blocks > 1024) blocks = 1024;

    edge_decoder_wmma_kernel<<<blocks, THREADS, 0, stream>>>(
        z_src, z_dst, eli, W1, b1, W2, b2, out, E, nTiles);
}
